// SimpleGATLayer_51032801411076
// MI455X (gfx1250) — matrix operand order repaired and checked by reading, not compiled, not run on hardware
//
#include <hip/hip_runtime.h>
#include <hip/hip_bf16.h>

typedef __attribute__((ext_vector_type(16))) _Float16 v16h;
typedef __attribute__((ext_vector_type(8)))  float    v8f;
typedef __attribute__((ext_vector_type(4)))  int      v4i;
typedef __attribute__((ext_vector_type(4)))  float    v4f;

#define Bb 8
#define Nn 2048
#define Ff 128
#define Uu 128
#define NEG_INF_V (-1000000000.0f)
#define LEAKY 0.2f
#define LDS_STRIDE 80                      // bytes per staged u-row (5*16B: aligned, bank-spread)
#define TILE_BYTES (128 * LDS_STRIDE)      // one staged 128u x 32j f16 tile (+pad)

// Issue 4 async 16B copies: one 64B u-row chunk, global -> LDS.
// IOFFSET applies to BOTH the LDS and the global address (ISA 08 §4.4),
// so a single LDS-addr VGPR + single 64-bit global-addr VGPR pair suffice.
#define ASYNC_CP_ROW64(ldsaddr, gptr)                                   \
  asm volatile(                                                         \
      "global_load_async_to_lds_b128 %0, %1, off\n\t"                   \
      "global_load_async_to_lds_b128 %0, %1, off offset:16\n\t"         \
      "global_load_async_to_lds_b128 %0, %1, off offset:32\n\t"         \
      "global_load_async_to_lds_b128 %0, %1, off offset:48"             \
      :: "v"(ldsaddr), "v"(gptr) : "memory")

// ---------------------------------------------------------------------------
// Kernel 1: H[b,n,u] = sum_f X[b,n,f] * W[f,u]  (f32)  +  HhT[b,u,n] (f16)
// ---------------------------------------------------------------------------
__global__ __launch_bounds__(128) void gat_h_kernel(
    const float* __restrict__ X, const float* __restrict__ W,
    float* __restrict__ H, _Float16* __restrict__ HhT) {
  const int tid = threadIdx.x;
  const int g0  = blockIdx.x * 16;       // global row base (b*N + n)
  const int b   = g0 / Nn;
  const int n0  = g0 % Nn;

  __shared__ float Xs[16][Ff];
  #pragma unroll
  for (int r = 0; r < 16; ++r)
    Xs[r][tid] = X[(size_t)(g0 + r) * Ff + tid];
  __syncthreads();

  float acc[16];
  #pragma unroll
  for (int r = 0; r < 16; ++r) acc[r] = 0.0f;

  for (int f = 0; f < Ff; ++f) {
    float w = W[f * Uu + tid];
    #pragma unroll
    for (int r = 0; r < 16; ++r) acc[r] = fmaf(Xs[r][f], w, acc[r]);
  }

  v16h hv;
  #pragma unroll
  for (int r = 0; r < 16; ++r) {
    H[(size_t)(g0 + r) * Uu + tid] = acc[r];
    hv[r] = (_Float16)acc[r];
  }
  *(v16h*)(HhT + ((size_t)(b * Uu + tid)) * Nn + n0) = hv;   // aligned 32B store
}

// ---------------------------------------------------------------------------
// Kernel 2: s[b,n] = H[b,n,:]·a_src ; t[b,n] = H[b,n,:]·a_dst
// ---------------------------------------------------------------------------
__global__ __launch_bounds__(128) void gat_st_kernel(
    const float* __restrict__ H, const float* __restrict__ a,
    float* __restrict__ s, float* __restrict__ t) {
  const int row = blockIdx.x;            // 0 .. B*N-1
  const int tid = threadIdx.x;
  float h  = H[(size_t)row * Uu + tid];
  float ps = h * a[tid];
  float pt = h * a[Uu + tid];
  #pragma unroll
  for (int off = 16; off > 0; off >>= 1) {
    ps += __shfl_down(ps, off, 32);
    pt += __shfl_down(pt, off, 32);
  }
  __shared__ float rs[4], rt[4];
  if ((tid & 31) == 0) { rs[tid >> 5] = ps; rt[tid >> 5] = pt; }
  __syncthreads();
  if (tid == 0) {
    s[row] = rs[0] + rs[1] + rs[2] + rs[3];
    t[row] = rt[0] + rt[1] + rt[2] + rt[3];
  }
}

// ---------------------------------------------------------------------------
// Kernel 3: online-softmax attention + (attn @ H) via v_wmma_f32_16x16x32_f16.
// 4 waves/block share one batch; H tiles double-buffered in LDS via
// global_load_async_to_lds_b128 (ASYNCcnt), overlapping the A-stream compute.
// ---------------------------------------------------------------------------
__global__ __launch_bounds__(128) void gat_attn_kernel(
    const int* __restrict__ A, const float* __restrict__ sv,
    const float* __restrict__ tv, const _Float16* __restrict__ HhT,
    float* __restrict__ out) {
  __shared__ char Bs[2][TILE_BYTES];     // 2 x 10 KiB staged H tiles

  const int tid  = threadIdx.x;
  const int lane = tid & 31;
  const int wave = tid >> 5;
  const int half = lane >> 4;            // 0 or 1
  const int l15  = lane & 15;
  const int gt   = blockIdx.x * 4 + wave;    // 4 consecutive i-tiles, same batch
  const int b    = (gt * 16) / Nn;
  const int i0   = (gt * 16) % Nn;
  const int i    = i0 + l15;             // this lane's stat row
  const int base = half * 8;             // first in-lane j offset (A-frag layout)

  const float s_i = sv[b * Nn + i];
  const int*   __restrict__ arow = A + ((size_t)(b * Nn + i)) * Nn;
  const float* __restrict__ trow = tv + b * Nn;
  // this thread's staging source: u-row (u = tid) of HhT[b]
  const _Float16* __restrict__ hrow = HhT + ((size_t)b * Uu + tid) * Nn;

  const unsigned lds0 = (unsigned)(unsigned long long)(void*)&Bs[0][0] + tid * LDS_STRIDE;
  const unsigned lds1 = lds0 + TILE_BYTES;

  float m = -3.0e38f;                    // running row max
  float l = 0.0f;                        // running row sum
  v8f acc[8];
  #pragma unroll
  for (int u = 0; u < 8; ++u)
    #pragma unroll
    for (int k = 0; k < 8; ++k) acc[u][k] = 0.0f;

  // prologue: stage chunk 0 into buffer 0
  ASYNC_CP_ROW64(lds0, (const void*)hrow);

  for (int jc = 0; jc < Nn; jc += 32) {
    const int cur = (jc >> 5) & 1;
    // stage next chunk into the other buffer (wraps on last iter so exactly
    // 4 asyncs stay in flight -> constant, deterministic wait threshold)
    {
      int jn = jc + 32; if (jn >= Nn) jn = 0;
      ASYNC_CP_ROW64(cur ? lds0 : lds1, (const void*)(hrow + jn));
    }

    // ---- load 16 adjacency ints + 16 t values (per A-fragment layout) ----
    v4i a0 = *(const v4i*)(arow + jc + base);
    v4i a1 = *(const v4i*)(arow + jc + base + 4);
    v4i a2 = *(const v4i*)(arow + jc + base + 16);
    v4i a3 = *(const v4i*)(arow + jc + base + 20);
    v4f t0 = *(const v4f*)(trow + jc + base);
    v4f t1 = *(const v4f*)(trow + jc + base + 4);
    v4f t2 = *(const v4f*)(trow + jc + base + 16);
    v4f t3 = *(const v4f*)(trow + jc + base + 20);
    if (jc + 32 < Nn) {
      __builtin_prefetch(arow + jc + 32 + base, 0, 0);       // global_prefetch_b8
      __builtin_prefetch(arow + jc + 32 + base + 16, 0, 0);
    }

    int   am[16]; float tj[16];
    #pragma unroll
    for (int q = 0; q < 4; ++q) {
      am[q]      = a0[q];  tj[q]      = t0[q];
      am[4 + q]  = a1[q];  tj[4 + q]  = t1[q];
      am[8 + q]  = a2[q];  tj[8 + q]  = t2[q];
      am[12 + q] = a3[q];  tj[12 + q] = t3[q];
    }

    // ---- masked leaky-relu logits; in-lane chunk max ----
    float e[16];
    float mx = -3.0e38f;
    #pragma unroll
    for (int k = 0; k < 16; ++k) {
      float x  = s_i + tj[k];
      float lr = (x > 0.0f) ? x : (LEAKY * x);
      float ev = lr + ((am[k] > 0) ? 0.0f : NEG_INF_V);
      e[k] = ev;
      mx = fmaxf(mx, ev);
    }

    // ---- online softmax statistics (row replicated across lane pair) ----
    float mrow = fmaxf(mx, __shfl_xor(mx, 16, 32));
    float newm = fmaxf(m, mrow);
    float alpha = __expf(m - newm);
    m = newm;

    float sl = 0.0f;
    v16h pf;
    #pragma unroll
    for (int k = 0; k < 16; ++k) {
      float p = __expf(e[k] - m);
      sl += p;
      pf[k] = (_Float16)p;
    }
    sl += __shfl_xor(sl, 16, 32);
    l = l * alpha + sl;

    // ---- rescale accumulators: alpha of row M=v+8*half fetched from lane M
    float av8[8];
    #pragma unroll
    for (int v = 0; v < 8; ++v) av8[v] = __shfl(alpha, half * 8 + v, 32);
    #pragma unroll
    for (int u = 0; u < 8; ++u)
      #pragma unroll
      for (int v = 0; v < 8; ++v) acc[u][v] *= av8[v];

    // ---- current tile ready: older 4 asyncs done (in-order), newest 4 may fly
    asm volatile("s_wait_asynccnt 0x4" ::: "memory");
    __syncthreads();                     // all 128 u-rows visible to all waves

    // ---- 8 u-tiles of the attn@H contraction via WMMA, B-frags from LDS ----
    const char* bbase = Bs[cur];
    #pragma unroll
    for (int ut = 0; ut < 8; ++ut) {
      typedef __attribute__((ext_vector_type(8))) _Float16 v8h_;
      v8h_ blo = *(const v8h_*)(bbase + (ut * 16 + l15) * LDS_STRIDE + half * 16);
      v8h_ bhi = *(const v8h_*)(bbase + (ut * 16 + l15) * LDS_STRIDE + 32 + half * 16);
      v16h bfrag = __builtin_shufflevector(blo, bhi, 0, 1, 2, 3, 4, 5, 6, 7, 8, 9, 10, 11, 12, 13, 14, 15);
      acc[ut] = __builtin_amdgcn_wmma_f32_16x16x32_f16(
          false, pf, false, bfrag, (short)0, acc[ut], false, false);
    }
    __syncthreads();                     // all waves done reading buf[cur]
  }

  // ---- finalize: divide by row sum, relu, store ----
  float lrow[8];
  #pragma unroll
  for (int v = 0; v < 8; ++v) lrow[v] = __shfl(l, half * 8 + v, 32);
  #pragma unroll
  for (int ut = 0; ut < 8; ++ut) {
    #pragma unroll
    for (int v = 0; v < 8; ++v) {
      float o = acc[ut][v] / lrow[v];
      o = fmaxf(o, 0.0f);
      out[((size_t)(b * Nn + i0 + half * 8 + v)) * Uu + ut * 16 + l15] = o;
    }
  }
}

// ---------------------------------------------------------------------------
extern "C" void kernel_launch(void* const* d_in, const int* in_sizes, int n_in,
                              void* d_out, int out_size, void* d_ws, size_t ws_size,
                              hipStream_t stream) {
  (void)in_sizes; (void)n_in; (void)out_size; (void)ws_size;
  const float* X = (const float*)d_in[0];
  const int*   A = (const int*)d_in[1];
  const float* W = (const float*)d_in[2];
  const float* a = (const float*)d_in[3];
  float* out = (float*)d_out;

  char* ws = (char*)d_ws;
  const size_t H_bytes   = (size_t)Bb * Nn * Uu * sizeof(float);     // 8 MiB
  const size_t HhT_bytes = (size_t)Bb * Uu * Nn * sizeof(_Float16);  // 4 MiB
  float*    H   = (float*)ws;
  _Float16* HhT = (_Float16*)(ws + H_bytes);
  float*    s   = (float*)(ws + H_bytes + HhT_bytes);
  float*    t   = s + (size_t)Bb * Nn;

  // K1: H = X@W (f32) + transposed f16 copy for WMMA B-operands
  gat_h_kernel<<<(Bb * Nn) / 16, 128, 0, stream>>>(X, W, H, HhT);
  // K2: s = H@a_src, t = H@a_dst
  gat_st_kernel<<<Bb * Nn, 128, 0, stream>>>(H, a, s, t);
  // K3: online-softmax attention + attn@H via v_wmma + async-LDS staging
  gat_attn_kernel<<<(Bb * Nn) / 64, 128, 0, stream>>>(A, s, t, HhT, out);
}
